// ScaledDotProductAttention_29996051595712
// MI455X (gfx1250) — matrix operand order repaired and checked by reading, not compiled, not run on hardware
//
#include <hip/hip_runtime.h>
#include <math.h>

// Problem constants (match reference)
#define B_  4
#define H_  16
#define S_  2048
#define DK_ 64
#define KB  64          // keys processed per block iteration

typedef _Float16 h2   __attribute__((ext_vector_type(2)));
typedef _Float16 h8   __attribute__((ext_vector_type(8)));
typedef _Float16 v16h __attribute__((ext_vector_type(16)));
typedef float    v8f  __attribute__((ext_vector_type(8)));

union V16 { v16h v; h8 h[2]; _Float16 e[16]; };
union C8  { v8f  v; float4 f[2]; float e[8]; };

// D = A(16x32 f16) * B(32x16 f16) + C(16x16 f32)
#define WMMA(A, Bm, C) __builtin_amdgcn_wmma_f32_16x16x32_f16( \
    false, (A), false, (Bm), (short)0, (C), false, false)

// LDS row strides in halfs: *2 bytes is a multiple of 16B (aligned b128)
// and not a multiple of 256B (bank spread).
#define KSTR 72         // sK row: 64 dk + 8 pad   -> 144 B
#define VSTR 72         // sV row: 64 key + 8 pad  -> 144 B

// 1/sqrt(64) * log2(e); folded into Q at f16-convert time (exp2-domain softmax)
#define SC 0.18033688011112042f

__launch_bounds__(256)
__global__ void attn_fwd_f16wmma(const float* __restrict__ q,
                                 const float* __restrict__ k,
                                 const float* __restrict__ v,
                                 const int*   __restrict__ mask,
                                 float*       __restrict__ out) {
  __shared__ __align__(16) _Float16 sK[KB * KSTR];    // [key_local][dk]  f16
  __shared__ __align__(16) _Float16 sV[DK_ * VSTR];   // [d][key_local]   f16 (transposed)
  __shared__ __align__(16) float    sbias[KB];        // 0.0f or -1e9f per key

  const int bh = blockIdx.y;                 // b*H + h
  const size_t base = (size_t)bh * S_ * DK_;
  const float* qb  = q + base;
  const float* kgb = k + base;
  const float* vgb = v + base;
  const int*   mp  = mask + (size_t)(bh / H_) * S_;
  float*       ob  = out + base;

  const int tid  = threadIdx.x;
  const int lane = tid & 31;
  const int wv   = tid >> 5;      // wave 0..7
  const int l16  = lane & 15;
  const int hi   = lane >> 4;     // lane half
  const int koff = hi * 8;        // A-operand / C-row offset for upper half
  const int runb = hi * 16;       // unused; B operands also offset by koff

  const int q0 = blockIdx.x * 256 + wv * 32;   // this wave's 32 query rows

  // ---- Q as B operands (Q^T), two 16-query tiles, scale folded in ----
  V16 bq[2][2];   // [qtile][lo/hi dk]
#pragma unroll
  for (int qt = 0; qt < 2; ++qt) {
    const float* qp = qb + (size_t)(q0 + 16 * qt + l16) * DK_;
#pragma unroll
    for (int j = 0; j < 16; ++j) {
      bq[qt][0].e[j] = (_Float16)(qp[koff + (j & 7) + 2 * (j & 8)]      * SC);
      bq[qt][1].e[j] = (_Float16)(qp[koff + (j & 7) + 2 * (j & 8) + 32] * SC);
    }
  }

  // Output accumulators: 2 qtiles x 4 d-tiles of 16x16
  v8f o[2][4];
#pragma unroll
  for (int qt = 0; qt < 2; ++qt)
#pragma unroll
    for (int i = 0; i < 4; ++i)
#pragma unroll
      for (int r = 0; r < 8; ++r) o[qt][i][r] = 0.0f;

  float m[2] = {-INFINITY, -INFINITY};
  float l[2] = {0.0f, 0.0f};

  // ---- staging work split (256 threads, KB=64 keys, DK=64) ----
  const int krow = tid >> 2;        // 0..63 : K row
  const int kseg = (tid & 3) * 16;  // 0/16/32/48 : K dk segment (16 floats)
  const int vp   = (tid & 31) * 2;  // key pair 0,2,..,62 for V transpose
  const int vseg = (tid >> 5) * 8;  // 0..56 : V d segment (8 floats x 2 keys)

  float4 kr[4], vr0[2], vr1[2];
  int mreg = 0;

#define ISSUE_LOADS(kb_)                                                       \
  {                                                                            \
    const float4* kg = (const float4*)(kgb + (size_t)((kb_) + krow) * DK_ + kseg); \
    kr[0] = kg[0]; kr[1] = kg[1]; kr[2] = kg[2]; kr[3] = kg[3];                \
    const float4* vg0 = (const float4*)(vgb + (size_t)((kb_) + vp) * DK_ + vseg);  \
    const float4* vg1 = (const float4*)(vgb + (size_t)((kb_) + vp + 1) * DK_ + vseg); \
    vr0[0] = vg0[0]; vr0[1] = vg0[1];                                          \
    vr1[0] = vg1[0]; vr1[1] = vg1[1];                                          \
    if (tid < KB) mreg = mp[(kb_) + tid];                                      \
  }

  ISSUE_LOADS(0);

  for (int kb = 0; kb < S_; kb += KB) {
    __syncthreads();   // previous iteration's LDS reads complete

    // ---- convert + store staged registers to LDS ----
    {
      const float* kf = (const float*)kr;
      h8 x0, x1;
#pragma unroll
      for (int j = 0; j < 8; ++j) {
        x0[j] = (_Float16)kf[j];
        x1[j] = (_Float16)kf[8 + j];
      }
      *(h8*)&sK[krow * KSTR + kseg]     = x0;   // ds_store_b128
      *(h8*)&sK[krow * KSTR + kseg + 8] = x1;

      const float* vf0 = (const float*)vr0;
      const float* vf1 = (const float*)vr1;
#pragma unroll
      for (int j = 0; j < 8; ++j) {             // packed key-pair stores
        h2 p; p[0] = (_Float16)vf0[j]; p[1] = (_Float16)vf1[j];
        *(h2*)&sV[(size_t)(vseg + j) * VSTR + vp] = p;
      }
      if (tid < KB) sbias[tid] = mreg ? 0.0f : -1.0e9f;
    }
    __syncthreads();

    // ---- issue next block's loads early: latency hides under compute ----
    if (kb + KB < S_) ISSUE_LOADS(kb + KB);
    if (kb + 2 * KB < S_) {
      __builtin_prefetch(kgb + (size_t)(kb + 2 * KB + krow) * DK_ + kseg, 0, 1);
      __builtin_prefetch(vgb + (size_t)(kb + 2 * KB + vp)   * DK_ + vseg, 0, 1);
    }

    // ---- process the two 16-query tiles sequentially ----
#pragma unroll
    for (int qt = 0; qt < 2; ++qt) {
      // S^T = K(64x64) x Q^T(64x16); C seeded with mask bias per key row
      C8 s[4];
#pragma unroll
      for (int t = 0; t < 4; ++t) {
        s[t].f[0] = *(const float4*)&sbias[16 * t + koff];     // bias seed
        s[t].f[1] = *(const float4*)&sbias[16 * t + koff + 4]; // (ds_load_b128)
        const _Float16* krt = &sK[(size_t)(16 * t + l16) * KSTR];
        V16 a;
        a.h[0] = *(const h8*)(krt + koff);
        a.h[1] = *(const h8*)(krt + 16 + koff);
        s[t].v = WMMA(a.v, bq[qt][0].v, s[t].v);
        a.h[0] = *(const h8*)(krt + 32 + koff);
        a.h[1] = *(const h8*)(krt + 48 + koff);
        s[t].v = WMMA(a.v, bq[qt][1].v, s[t].v);
      }

      // online softmax (exp2 domain, rows lane-local + one lane-pair combine)
      float mb = -INFINITY;
#pragma unroll
      for (int t = 0; t < 4; ++t)
#pragma unroll
        for (int r = 0; r < 8; ++r) mb = fmaxf(mb, s[t].e[r]);
      mb = fmaxf(mb, __shfl_xor(mb, 16, 32));

      const bool need  = (__ballot(mb > m[qt]) != 0ull);  // wave-uniform
      const float mn    = fmaxf(m[qt], mb);
      const float alpha = exp2f(m[qt] - mn);

      V16 pa0, pa1;   // P as A operands: keys 0..31 / 32..63
      float lb = 0.0f;
#pragma unroll
      for (int r = 0; r < 8; ++r) {
        float p0 = exp2f(s[0].e[r] - mn);
        float p1 = exp2f(s[1].e[r] - mn);
        float p2 = exp2f(s[2].e[r] - mn);
        float p3 = exp2f(s[3].e[r] - mn);
        lb += (p0 + p1) + (p2 + p3);
        pa0.e[r] = (_Float16)p0; pa0.e[8 + r] = (_Float16)p1;
        pa1.e[r] = (_Float16)p2; pa1.e[8 + r] = (_Float16)p3;
      }
      lb += __shfl_xor(lb, 16, 32);
      l[qt] = l[qt] * alpha + lb;
      m[qt] = mn;

      if (need) {   // rescale O only when the running max actually moved
#pragma unroll
        for (int r = 0; r < 8; ++r) {
          const float ar = __shfl(alpha, r + koff, 32);
#pragma unroll
          for (int i = 0; i < 4; ++i) o[qt][i][r] *= ar;
        }
      }

      // O += P(16x64) x V(64x64): 8 WMMAs
#pragma unroll
      for (int i = 0; i < 4; ++i) {
        const _Float16* vrow = &sV[(size_t)(16 * i + l16) * VSTR];
        V16 vb;
        vb.h[0] = *(const h8*)(vrow + koff);
        vb.h[1] = *(const h8*)(vrow + 16 + koff);
        o[qt][i] = WMMA(pa0.v, vb.v, o[qt][i]);
        vb.h[0] = *(const h8*)(vrow + 32 + koff);
        vb.h[1] = *(const h8*)(vrow + 48 + koff);
        o[qt][i] = WMMA(pa1.v, vb.v, o[qt][i]);
      }
    }
  }

  // ---- normalize by 1/l and store ----
#pragma unroll
  for (int qt = 0; qt < 2; ++qt) {
#pragma unroll
    for (int r = 0; r < 8; ++r) {
      const float linv = 1.0f / __shfl(l[qt], r + koff, 32);
      float* orow = ob + (size_t)(q0 + 16 * qt + koff + r) * DK_ + l16;
#pragma unroll
      for (int i = 0; i < 4; ++i) orow[16 * i] = o[qt][i][r] * linv;
    }
  }
}

extern "C" void kernel_launch(void* const* d_in, const int* in_sizes, int n_in,
                              void* d_out, int out_size, void* d_ws, size_t ws_size,
                              hipStream_t stream) {
  const float* q    = (const float*)d_in[0];
  const float* k    = (const float*)d_in[1];
  const float* v    = (const float*)d_in[2];
  const int*   mask = (const int*)d_in[3];
  float*       out  = (float*)d_out;

  dim3 grid(S_ / 256, B_ * H_);  // (8, 64): 256 queries per WG
  dim3 block(256);               // 8 waves of 32
  attn_fwd_f16wmma<<<grid, block, 0, stream>>>(q, k, v, mask, out);
}
